// downprompt_10316511445589
// MI455X (gfx1250) — matrix operand order repaired and checked by reading, not compiled, not run on hardware
//
#include <hip/hip_runtime.h>
#include <hip/hip_bf16.h>
#include <cmath>

typedef __attribute__((ext_vector_type(16))) __bf16 v16bf;
typedef __attribute__((ext_vector_type(8)))  __bf16 v8bf;
typedef __attribute__((ext_vector_type(8)))  float  v8f;

#define DDIM 256
#define NB_CLASSES 10
#define EPSV 1e-8f

// ---------------- elementwise helpers ----------------

__global__ void k_fill(float* __restrict__ p, float v, long n) {
  long i = (long)blockIdx.x * blockDim.x + threadIdx.x;
  if (i < n) p[i] = v;
}

__global__ void k_deg(const int* __restrict__ dst, float* __restrict__ deg, int E) {
  int i = blockIdx.x * blockDim.x + threadIdx.x;
  if (i < E) unsafeAtomicAdd(&deg[dst[i]], 1.0f);
}

__global__ void k_dinv(float* __restrict__ dinv, float* __restrict__ dinv2, int N) {
  int i = blockIdx.x * blockDim.x + threadIdx.x;
  if (i < N) {
    float d = dinv[i];           // holds degree (incl. self loop)
    dinv[i]  = rsqrtf(d);        // D^-1/2
    dinv2[i] = 1.0f / d;         // self-loop coefficient
  }
}

__global__ void k_convert_bf16(const float* __restrict__ in, __bf16* __restrict__ out,
                               long n) {
  long i = (long)blockIdx.x * blockDim.x + threadIdx.x;
  if (i < n) out[i] = (__bf16)in[i];
}

// W is [K, Nc] row-major f32; Wt is [Nc, K] row-major bf16 (B-operand, K contiguous)
__global__ void k_transpose_bf16(const float* __restrict__ W, __bf16* __restrict__ Wt,
                                 int K, int Nc) {
  int i = blockIdx.x * blockDim.x + threadIdx.x;
  if (i < K * Nc) {
    int k = i / Nc, n = i - k * Nc;
    Wt[(long)n * K + k] = (__bf16)W[i];
  }
}

// ---------------- WMMA bf16 GEMM: C[M,Nc] = A[M,K] * Bt[Nc,K]^T ----------------
// One wave per 16x64 output tile: the A fragment (2 b128 loads / K-step) feeds
// 4 v_wmma_f32_16x16x32_bf16 into 4 accumulators -> 4x A-operand register reuse.
// Epilogue: optional bias, optional ELU, optional elementwise multiply by aux,
// and optional fused bf16 emission of the result (obf) for the next GEMM.
__global__ void k_gemm_wmma(const __bf16* __restrict__ A, const __bf16* __restrict__ Bt,
                            float* __restrict__ C, int Mrows, int K, int Nc,
                            const float* __restrict__ bias, int act,
                            const float* __restrict__ mulaux,
                            __bf16* __restrict__ obf) {
  int gid  = blockIdx.x * blockDim.x + threadIdx.x;
  int wave = gid >> 5;                    // wave32
  int lane = threadIdx.x & 31;
  int tiles_n = Nc >> 6;                  // 64-wide N tiles
  int tiles   = (Mrows >> 4) * tiles_n;
  if (wave >= tiles) return;              // wave-uniform: EXEC all-1s inside
  int tm  = wave / tiles_n;
  int tn4 = wave - tm * tiles_n;
  int row = lane & 15;
  int hi  = lane >> 4;

  // 16-bit A layout: lanes 0-15 hold K = {0..7, 16..23}; lanes 16-31 K = {8..15, 24..31}
  const __bf16* ap = A  + (size_t)(tm * 16 + row) * K + hi * 8;
  // 16-bit B layout: lanes 0-15 hold K = {0..7, 16..23}; lanes 16-31 K = {8..15, 24..31}
  const __bf16* bp = Bt + (size_t)(tn4 * 64 + row) * K + hi * 8;

  v8f acc[4] = {};
  for (int k0 = 0; k0 < K; k0 += 32) {
    v8bf a0 = *(const v8bf*)(ap + k0);
    v8bf a1 = *(const v8bf*)(ap + k0 + 16);
    v16bf a = __builtin_shufflevector(a0, a1, 0,1,2,3,4,5,6,7,8,9,10,11,12,13,14,15);
#pragma unroll
    for (int j = 0; j < 4; ++j) {
      const __bf16* bj = bp + (size_t)j * 16 * K;
      v8bf b0 = *(const v8bf*)(bj + k0);
      v8bf b1 = *(const v8bf*)(bj + k0 + 16);
      v16bf b = __builtin_shufflevector(b0, b1, 0,1,2,3,4,5,6,7,8,9,10,11,12,13,14,15);
      acc[j] = __builtin_amdgcn_wmma_f32_16x16x32_bf16(false, a, false, b, (short)0,
                                                       acc[j], false, false);
    }
  }

  // C/D layout: lane 0-15 -> N=lane, M=vgpr; lane 16-31 -> N=lane-16, M=8+vgpr
#pragma unroll
  for (int j = 0; j < 4; ++j) {
    int col = tn4 * 64 + j * 16 + row;
    float bv = bias ? bias[col] : 0.0f;
#pragma unroll
    for (int i = 0; i < 8; ++i) {
      size_t r = (size_t)(tm * 16 + hi * 8 + i);
      float v = acc[j][i] + bv;
      if (act) v = v > 0.0f ? v : expm1f(v);
      if (mulaux) v *= mulaux[r * Nc + col];
      C[r * Nc + col] = v;
      if (obf) obf[r * Nc + col] = (__bf16)v;
    }
  }
}

// ---------------- GCN message passing ----------------
// one block per edge, thread = feature; coalesced gather + hw f32 atomic scatter
__global__ void k_agg(const int* __restrict__ src, const int* __restrict__ dst,
                      const float* __restrict__ dinv, const float* __restrict__ h,
                      float* __restrict__ agg) {
  int e = blockIdx.x, t = threadIdx.x;
  int s = src[e], d = dst[e];
  float coef = dinv[s] * dinv[d];
  unsafeAtomicAdd(&agg[(size_t)d * DDIM + t], h[(size_t)s * DDIM + t] * coef);
}

// out = agg + h * (1/deg) + bias [+ residual] [elu]; optional fused bf16 emission
// of (out * scale) for the next GEMM's A operand (scale = gcn_weight2 or 1).
__global__ void k_combine(const float* __restrict__ agg, const float* __restrict__ h,
                          const float* __restrict__ dinv2, const float* __restrict__ bias,
                          const float* __restrict__ res, float* __restrict__ out,
                          int act, __bf16* __restrict__ obf,
                          const float* __restrict__ oscale, long n) {
  long i = (long)blockIdx.x * blockDim.x + threadIdx.x;
  if (i < n) {
    long node = i >> 8;
    int  d    = (int)(i & 255);
    float v = agg[i] + h[i] * dinv2[node] + bias[d];
    if (res) v += res[i];
    if (act) v = v > 0.0f ? v : expm1f(v);
    out[i] = v;
    if (obf) {
      float s = oscale ? oscale[0] : 1.0f;
      obf[i] = (__bf16)(v * s);
    }
  }
}

// ---------------- prompt attention + gather ----------------
// block m: node = idx[m]; softmax over 5 prompt logits; rawret[m] = h + w @ p_list
__global__ void k_attention(const float* __restrict__ h, const int* __restrict__ idx,
                            const float* __restrict__ aW, const float* __restrict__ ab,
                            const float* __restrict__ plist, float* __restrict__ rawret) {
  __shared__ float sh[256];
  __shared__ float slog[5];
  int m = blockIdx.x, t = threadIdx.x;
  int node = idx[m];
  float hv = h[(size_t)node * DDIM + t];
  for (int j = 0; j < 5; ++j) {
    sh[t] = hv * aW[t * 5 + j];
    __syncthreads();
    for (int s = 128; s > 0; s >>= 1) { if (t < s) sh[t] += sh[t + s]; __syncthreads(); }
    if (t == 0) slog[j] = sh[0] + ab[j];
    __syncthreads();
  }
  float mx = slog[0];
  for (int j = 1; j < 5; ++j) mx = fmaxf(mx, slog[j]);
  float w[5], se = 0.0f;
  for (int j = 0; j < 5; ++j) { w[j] = __expf(slog[j] - mx); se += w[j]; }
  float e = hv;
  for (int j = 0; j < 5; ++j) e += (w[j] / se) * plist[j * DDIM + t];
  rawret[(size_t)m * DDIM + t] = e;
}

__global__ void k_proto_accum(const float* __restrict__ rawret, const int* __restrict__ labels,
                              float* __restrict__ sums, float* __restrict__ cnts) {
  int m = blockIdx.x, t = threadIdx.x;
  int c = labels[m];
  unsafeAtomicAdd(&sums[(size_t)c * DDIM + t], rawret[(size_t)m * DDIM + t]);
  if (t == 0) unsafeAtomicAdd(&cnts[c], 1.0f);
}

__global__ void k_proto_norm(const float* __restrict__ sums, const float* __restrict__ cnts,
                             float* __restrict__ an) {
  __shared__ float sh[256];
  __shared__ float snorm;
  int c = blockIdx.x, t = threadIdx.x;
  float a = sums[(size_t)c * DDIM + t] / fmaxf(cnts[c], 1.0f);
  sh[t] = a * a;
  __syncthreads();
  for (int s = 128; s > 0; s >>= 1) { if (t < s) sh[t] += sh[t + s]; __syncthreads(); }
  if (t == 0) snorm = fmaxf(sqrtf(sh[0]), EPSV);
  __syncthreads();
  an[(size_t)c * DDIM + t] = a / snorm;
}

// cosine similarity + softmax over NB_CLASSES
__global__ void k_final(const float* __restrict__ rawret, const float* __restrict__ an,
                        float* __restrict__ out) {
  __shared__ float sh[256];
  __shared__ float sdots[NB_CLASSES];
  __shared__ float snorm;
  int m = blockIdx.x, t = threadIdx.x;
  float r = rawret[(size_t)m * DDIM + t];
  sh[t] = r * r;
  __syncthreads();
  for (int s = 128; s > 0; s >>= 1) { if (t < s) sh[t] += sh[t + s]; __syncthreads(); }
  if (t == 0) snorm = fmaxf(sqrtf(sh[0]), EPSV);
  __syncthreads();
  for (int c = 0; c < NB_CLASSES; ++c) {
    sh[t] = r * an[(size_t)c * DDIM + t];
    __syncthreads();
    for (int s = 128; s > 0; s >>= 1) { if (t < s) sh[t] += sh[t + s]; __syncthreads(); }
    if (t == 0) sdots[c] = sh[0] / snorm;
    __syncthreads();
  }
  if (t == 0) {
    float mx = sdots[0];
    for (int c = 1; c < NB_CLASSES; ++c) mx = fmaxf(mx, sdots[c]);
    float se = 0.0f, ev[NB_CLASSES];
    for (int c = 0; c < NB_CLASSES; ++c) { ev[c] = __expf(sdots[c] - mx); se += ev[c]; }
    for (int c = 0; c < NB_CLASSES; ++c) out[(size_t)m * NB_CLASSES + c] = ev[c] / se;
  }
}

// ---------------- host orchestration ----------------

extern "C" void kernel_launch(void* const* d_in, const int* in_sizes, int n_in,
                              void* d_out, int out_size, void* d_ws, size_t ws_size,
                              hipStream_t stream) {
  (void)n_in; (void)out_size; (void)ws_size;
  const int D = DDIM;
  const int N = in_sizes[0] / D;
  const int E = in_sizes[1];
  const int M = in_sizes[3];
  const int T = in_sizes[13] / D;   // cn_b_in is (T, H=256)
  const size_t ND = (size_t)N * D;

  const float* x_in   = (const float*)d_in[0];
  const int*   srcp   = (const int*)d_in[1];
  const int*   dstp   = (const int*)d_in[2];
  const int*   idxp   = (const int*)d_in[3];
  const int*   labels = (const int*)d_in[4];
  const float* W1     = (const float*)d_in[6];
  const float* b1     = (const float*)d_in[7];
  const float* W2     = (const float*)d_in[8];
  const float* b2     = (const float*)d_in[9];
  const float* W3     = (const float*)d_in[10];
  const float* b3     = (const float*)d_in[11];
  const float* cwin   = (const float*)d_in[12];
  const float* cbin   = (const float*)d_in[13];
  const float* cwhid  = (const float*)d_in[14];
  const float* cbhid  = (const float*)d_in[15];
  const float* cwout  = (const float*)d_in[16];
  const float* cbout  = (const float*)d_in[17];
  const float* plist  = (const float*)d_in[18];
  const float* aW     = (const float*)d_in[19];
  const float* ab     = (const float*)d_in[20];
  const float* w2scl  = (const float*)d_in[21];   // gcn_weight2 scalar (device)

  // workspace carve (256B aligned)
  char* p = (char*)d_ws;
  auto carve = [&](size_t bytes) { char* r = p; p += (bytes + 255) & ~(size_t)255; return r; };
  float* dinv   = (float*)carve((size_t)N * 4);
  float* dinv2  = (float*)carve((size_t)N * 4);
  float* bufX   = (float*)carve(ND * 4);
  float* bufH   = (float*)carve(ND * 4);
  float* bufAgg = (float*)carve(ND * 4);
  float* bufE1  = (float*)carve(ND * 4);
  float* rawret = (float*)carve((size_t)M * D * 4);
  float* sums   = (float*)carve((size_t)NB_CLASSES * D * 4);
  float* cnts   = (float*)carve((size_t)NB_CLASSES * 4);
  float* anb    = (float*)carve((size_t)NB_CLASSES * D * 4);
  __bf16* xbf[2];
  xbf[0] = (__bf16*)carve(ND * 2);
  xbf[1] = (__bf16*)carve(ND * 2);
  __bf16* wt[12];
  for (int i = 0; i < 12; ++i) wt[i] = (__bf16*)carve((size_t)D * D * 2);
  // wt[0..2]=W1t,W2t,W3t ; wt[3+t]=w_in[t]^T ; wt[6+t]=w_hid[t]^T ; wt[9+t]=w_out[t]^T

  const int TPB = 256;
  const int gND = (int)((ND + TPB - 1) / TPB);
  const int gE  = (E + TPB - 1) / TPB;
  const int gN  = (N + TPB - 1) / TPB;
  const int gW  = (D * D + TPB - 1) / TPB;
  const int gemmBlocks = (int)(((size_t)(N / 16) * (D / 64) * 32 + TPB - 1) / TPB);

  // degrees (with self loop), D^-1/2 and 1/deg
  k_fill<<<gN, TPB, 0, stream>>>(dinv, 1.0f, (long)N);
  k_deg<<<gE, TPB, 0, stream>>>(dstp, dinv, E);
  k_dinv<<<gN, TPB, 0, stream>>>(dinv, dinv2, N);

  // pre-transpose all weights to bf16 (K-contiguous B operands)
  k_transpose_bf16<<<gW, TPB, 0, stream>>>(W1, wt[0], D, D);
  k_transpose_bf16<<<gW, TPB, 0, stream>>>(W2, wt[1], D, D);
  k_transpose_bf16<<<gW, TPB, 0, stream>>>(W3, wt[2], D, D);
  for (int t = 0; t < T && t < 3; ++t) {
    k_transpose_bf16<<<gW, TPB, 0, stream>>>(cwin  + (size_t)t * D * D, wt[3 + t], D, D);
    k_transpose_bf16<<<gW, TPB, 0, stream>>>(cwhid + (size_t)t * D * D, wt[6 + t], D, D);
    k_transpose_bf16<<<gW, TPB, 0, stream>>>(cwout + (size_t)t * D * D, wt[9 + t], D, D);
  }

  // bf16 of the initial x; thereafter bf16 staging is produced by fused epilogues
  int cur = 0;
  k_convert_bf16<<<gND, TPB, 0, stream>>>(x_in, xbf[0], (long)ND);

  // gcn_conv: A = xbf[cur] (bf16 of conv input); writes out f32 and bf16(out*oscale)
  auto gcn = [&](const __bf16* Wt, const float* bias, float* out,
                 const float* res, int act, const float* oscale) {
    k_gemm_wmma<<<gemmBlocks, TPB, 0, stream>>>(xbf[cur], Wt, bufH, N, D, D,
                                                nullptr, 0, nullptr, nullptr);
    k_fill<<<gND, TPB, 0, stream>>>(bufAgg, 0.0f, (long)ND);
    k_agg<<<E, TPB, 0, stream>>>(srcp, dstp, dinv, bufH, bufAgg);
    k_combine<<<gND, TPB, 0, stream>>>(bufAgg, bufH, dinv2, bias, res, out, act,
                                       xbf[cur ^ 1], oscale, (long)ND);
    cur ^= 1;
  };

  // think loop: e3 is dead (w1 = w3 = 0), embed = gcn_weight2 * e2
  for (int t = 0; t < T; ++t) {
    gcn(wt[0], b1, bufE1, nullptr, 0, nullptr);            // e1 (+ bf16(e1))
    gcn(wt[1], b2, bufX,  bufE1,  0, w2scl);               // e2 (+ bf16(e2*w2))
    // ConditionNet: elu(embed@win+bin) -> elu(.@whid+bhid) -> .@wout+bout; x = prompt*origin
    k_gemm_wmma<<<gemmBlocks, TPB, 0, stream>>>(xbf[cur], wt[3 + t], bufH, N, D, D,
                                                cbin + (size_t)t * D, 1, nullptr,
                                                xbf[cur ^ 1]);
    cur ^= 1;
    k_gemm_wmma<<<gemmBlocks, TPB, 0, stream>>>(xbf[cur], wt[6 + t], bufAgg, N, D, D,
                                                cbhid + (size_t)t * D, 1, nullptr,
                                                xbf[cur ^ 1]);
    cur ^= 1;
    k_gemm_wmma<<<gemmBlocks, TPB, 0, stream>>>(xbf[cur], wt[9 + t], bufX, N, D, D,
                                                cbout + (size_t)t * D, 0, x_in,
                                                xbf[cur ^ 1]);   // bf16(new x)
    cur ^= 1;
  }

  // final GCN forward with ELU (bf16 of each layer fused for the next conv)
  gcn(wt[0], b1, bufE1, nullptr, 1, nullptr);
  gcn(wt[1], b2, bufE1, nullptr, 1, nullptr);
  gcn(wt[2], b3, bufE1, nullptr, 1, nullptr);

  // prompt attention on the M gathered nodes, prototypes, cosine softmax
  k_attention<<<M, TPB, 0, stream>>>(bufE1, idxp, aW, ab, plist, rawret);
  k_fill<<<(NB_CLASSES * D + TPB - 1) / TPB, TPB, 0, stream>>>(sums, 0.0f,
                                                               (long)NB_CLASSES * D);
  k_fill<<<1, TPB, 0, stream>>>(cnts, 0.0f, (long)NB_CLASSES);
  k_proto_accum<<<M, TPB, 0, stream>>>(rawret, labels, sums, cnts);
  k_proto_norm<<<NB_CLASSES, TPB, 0, stream>>>(sums, cnts, anb);
  k_final<<<M, TPB, 0, stream>>>(rawret, anb, (float*)d_out);
}
